// HighResDINO_996432413272
// MI455X (gfx1250) — matrix operand order repaired and checked by reading, not compiled, not run on hardware
//
#include <hip/hip_runtime.h>
#include <cstddef>

// ---------------------------------------------------------------------------
// ViT-S (D=384, H=6, L=12) forward with thresholded-softmax attention for
// MI455X / gfx1250. All matmuls run through v_wmma_f32_16x16x32_bf16.
//
// Softmax masking identity: attn_i >= RATIO*attn_max  <=>  s_i >= s_max+ln(RATIO)
// so threshold + renormalize folds into one masked exp-sum pass.
//
// Padding: tokens padded 962->976 (Np), rows padded 1924->1936 (Mp).
// qkv-unpack zero-fills padded Q/K/VT so the padded K-dimension of the
// attn@V GEMM contributes exactly 0 (avoids 0*NaN). Padded M rows only ever
// produce garbage in their own C rows (WMMA rows are independent) and all
// C stores are guarded by Mvalid.
// ---------------------------------------------------------------------------

typedef __attribute__((ext_vector_type(16))) __bf16 bf16x16;
typedef __attribute__((ext_vector_type(8)))  __bf16 bf16x8;
typedef __attribute__((ext_vector_type(8)))  float  f32x8;

#define DIM   384
#define HEADS 6
#define LAYERS 12
#define BATCH 2
#define NTOK  962
#define NP    976           // padded tokens (61*16)
#define MROWS (BATCH*NTOK)  // 1924
#define MP    1936          // padded rows (121*16)
#define DH    64
#define D3    1152
#define D4    1536

// ---------------------------------------------------------------------------
// Generic batched WMMA GEMM:  C[M x N] = A[M x K] * B[K x N] (+bias)(gelu)(+res)
// B supplied pre-transposed: Bt[N x K] row-major (bf16).
// One wave per block; each wave computes one 16(M) x 64(N) strip.
//
// The k-loop is branch-free: out-of-range N tiles are handled by clamping the
// B row index (loads stay in-bounds) and discarding at the store guard. This
// keeps the accumulators pinned (no phi-induced v_mov storms) and EXEC all-1s.
//
// Operand layouts used here (16-bit, wave32; contiguous k, same for A and B, not the ISA table order):
//   A 16x32 : lanes 0-15 row M=lane, halves 0..15=K0..15 contiguous;
//             lanes 16-31 same rows, K=16..31    -> two 16B loads per lane.
//   B 32x16 : lane n (0-15) holds column N=n, K=0..15 contiguous;
//             lanes 16-31 hold K=16..31          -> one 32B load from Bt[n][k].
// ---------------------------------------------------------------------------
template<bool BIAS, bool GELU, bool RES, bool STF, bool STB>
__global__ __launch_bounds__(32)
void gemm_wmma_kernel(const __bf16* __restrict__ A, int lda, size_t sA,
                      const __bf16* __restrict__ Bt, int ldb, size_t sB,
                      int K, int Mvalid, int Nvalid,
                      const float* __restrict__ bias,
                      const float* __restrict__ res, int ldr,
                      float* __restrict__ Cf, int ldc, size_t sC,
                      __bf16* __restrict__ Cb, int ldcb, size_t sCb)
{
    const int lane = threadIdx.x & 31;
    const int hl   = lane >> 4;    // 0: lanes 0-15, 1: lanes 16-31
    const int l16  = lane & 15;
    const int m0   = blockIdx.y * 16;
    const int n0   = blockIdx.x * 64;
    const int z    = blockIdx.z;

    A  += (size_t)z * sA;
    Bt += (size_t)z * sB;

    // Clamped B row pointers: all 4 tiles always load in-bounds data so the
    // inner loop stays straight-line; invalid tiles are dropped at the store.
    const __bf16* bp[4];
#pragma unroll
    for (int j = 0; j < 4; ++j) {
        int n = n0 + j * 16 + l16;
        if (n >= Nvalid) n = Nvalid - 1;
        bp[j] = Bt + (size_t)n * ldb + hl * 16;
    }

    union AFrag { bf16x16 v; bf16x8 h[2]; };
    f32x8 acc[4] = {};
    const __bf16* ap = A + (size_t)(m0 + l16) * lda + hl * 16;

    for (int k0 = 0; k0 < K; k0 += 32) {
        AFrag a;
        a.h[0] = *(const bf16x8*)(ap + k0);
        a.h[1] = *(const bf16x8*)(ap + k0 + 8);
        bf16x16 b0 = *(const bf16x16*)(bp[0] + k0);
        bf16x16 b1 = *(const bf16x16*)(bp[1] + k0);
        bf16x16 b2 = *(const bf16x16*)(bp[2] + k0);
        bf16x16 b3 = *(const bf16x16*)(bp[3] + k0);
        acc[0] = __builtin_amdgcn_wmma_f32_16x16x32_bf16(
            false, a.v, false, b0, (short)0, acc[0], false, false);
        acc[1] = __builtin_amdgcn_wmma_f32_16x16x32_bf16(
            false, a.v, false, b1, (short)0, acc[1], false, false);
        acc[2] = __builtin_amdgcn_wmma_f32_16x16x32_bf16(
            false, a.v, false, b2, (short)0, acc[2], false, false);
        acc[3] = __builtin_amdgcn_wmma_f32_16x16x32_bf16(
            false, a.v, false, b3, (short)0, acc[3], false, false);
    }

#pragma unroll
    for (int j = 0; j < 4; ++j) {
        const int col = n0 + j * 16 + l16;
        if (col >= Nvalid) continue;
        float bval = 0.0f;
        if (BIAS) bval = bias[col];
#pragma unroll
        for (int r = 0; r < 8; ++r) {
            const int row = m0 + r + hl * 8;  // C layout: VGPR r -> M = r + 8*hl
            if (row >= Mvalid) continue;
            float v = acc[j][r];
            if (BIAS) v += bval;
            if (GELU) v = 0.5f * v * (1.0f + erff(v * 0.70710678118654752f));
            if (RES)  v += res[(size_t)row * ldr + col];
            if (STF)  Cf[(size_t)z * sC  + (size_t)row * ldc  + col] = v;
            if (STB)  Cb[(size_t)z * sCb + (size_t)row * ldcb + col] = (__bf16)v;
        }
    }
}

// ---------------------------------------------------------------------------
// LayerNorm, one wave per row of 384 (12 elems/lane), shfl_xor reductions.
// ---------------------------------------------------------------------------
template<bool OUTF, bool OUTB>
__global__ __launch_bounds__(128)
void ln_kernel(const float* __restrict__ x, int rows,
               const float* __restrict__ g, const float* __restrict__ b,
               float* __restrict__ of, __bf16* __restrict__ ob)
{
    const int wid = threadIdx.x >> 5;
    const int lane = threadIdx.x & 31;
    const int row = blockIdx.x * 4 + wid;
    if (row >= rows) return;
    const float* xr = x + (size_t)row * DIM;

    float vals[12];
    float s = 0.0f;
#pragma unroll
    for (int i = 0; i < 12; ++i) { vals[i] = xr[lane + i * 32]; s += vals[i]; }
#pragma unroll
    for (int off = 16; off; off >>= 1) s += __shfl_xor(s, off, 32);
    const float mean = s * (1.0f / DIM);

    float ss = 0.0f;
#pragma unroll
    for (int i = 0; i < 12; ++i) { float d = vals[i] - mean; ss += d * d; }
#pragma unroll
    for (int off = 16; off; off >>= 1) ss += __shfl_xor(ss, off, 32);
    const float inv = rsqrtf(ss * (1.0f / DIM) + 1e-6f);

#pragma unroll
    for (int i = 0; i < 12; ++i) {
        const int c = lane + i * 32;
        const float y = (vals[i] - mean) * inv * g[c] + b[c];
        if (OUTF) of[(size_t)row * DIM + c] = y;
        if (OUTB) ob[(size_t)row * DIM + c] = (__bf16)y;
    }
}

// ---------------------------------------------------------------------------
// Masked softmax: keep s >= s_max + ln(0.1), renormalize over kept entries.
// One wave per query row. Writes bf16 probs (padded cols zeroed) and,
// on the last layer, f32 probs to d_out.
// ---------------------------------------------------------------------------
__global__ __launch_bounds__(128)
void softmax_mask_kernel(const float* __restrict__ scores,
                         __bf16* __restrict__ attnb,
                         float* __restrict__ attnf, int nrows)
{
    const int wid = threadIdx.x >> 5;
    const int lane = threadIdx.x & 31;
    const int r = blockIdx.x * 4 + wid;
    if (r >= nrows) return;                 // nrows = B*H*962
    const int bh = r / NTOK, q = r % NTOK;
    const float* sr = scores + ((size_t)bh * NP + q) * NP;

    float m = -3.0e38f;
    for (int c = lane; c < NTOK; c += 32) m = fmaxf(m, sr[c]);
#pragma unroll
    for (int off = 16; off; off >>= 1) m = fmaxf(m, __shfl_xor(m, off, 32));

    const float thr = m - 2.3025850929940457f;  // + ln(0.1)
    float s = 0.0f;
    for (int c = lane; c < NTOK; c += 32) {
        const float sc = sr[c];
        if (sc >= thr) s += expf(sc - m);
    }
#pragma unroll
    for (int off = 16; off; off >>= 1) s += __shfl_xor(s, off, 32);
    const float invs = 1.0f / s;

    __bf16* ab = attnb + ((size_t)bh * NP + q) * NP;
    float* af = attnf ? attnf + ((size_t)bh * NTOK + q) * NTOK : (float*)0;
    for (int c = lane; c < NP; c += 32) {
        float p = 0.0f;
        if (c < NTOK) {
            const float sc = sr[c];
            if (sc >= thr) p = expf(sc - m) * invs;
        }
        ab[c] = (__bf16)p;
        if (af && c < NTOK) af[c] = p;
    }
}

// ---------------------------------------------------------------------------
// Unpack qkv[M x 1152] -> Q,K bf16 [BH][Np][64] (Q pre-scaled by dh^-0.5),
// VT bf16 [BH][64][Np]. Zero-fills padded tokens (finite -> no NaN in PV GEMM).
// ---------------------------------------------------------------------------
__global__ __launch_bounds__(256)
void unpack_qkv_kernel(const float* __restrict__ qkv,
                       __bf16* __restrict__ Q, __bf16* __restrict__ Kd,
                       __bf16* __restrict__ VT)
{
    const int idx = blockIdx.x * 256 + threadIdx.x;
    const int total = BATCH * HEADS * NP * DH;
    if (idx >= total) return;
    const int d  = idx & 63;
    const int n  = (idx >> 6) % NP;
    const int bh = idx / (NP * DH);
    const int b  = bh / HEADS, h = bh % HEADS;

    float qv = 0.0f, kv = 0.0f, vv = 0.0f;
    if (n < NTOK) {
        const float* src = qkv + (size_t)(b * NTOK + n) * D3 + h * DH + d;
        qv = src[0] * 0.125f;    // dh^-0.5 folded into Q
        kv = src[DIM];
        vv = src[2 * DIM];
    }
    Q [((size_t)bh * NP + n) * DH + d] = (__bf16)qv;
    Kd[((size_t)bh * NP + n) * DH + d] = (__bf16)kv;
    VT[((size_t)bh * DH + d) * NP + n] = (__bf16)vv;
}

// W[K x N] f32 -> WT[N x K] bf16
__global__ __launch_bounds__(256)
void transpose_to_bf16_kernel(const float* __restrict__ W, int K, int N,
                              __bf16* __restrict__ WT)
{
    const long long idx = (long long)blockIdx.x * 256 + threadIdx.x;
    if (idx >= (long long)K * N) return;
    const int n = (int)(idx % N);
    const int k = (int)(idx / N);
    WT[(size_t)n * K + k] = (__bf16)W[(size_t)k * N + n];
}

// ---------------------------------------------------------------------------
// Host side
// ---------------------------------------------------------------------------
static inline size_t align256(size_t v) { return (v + 255) & ~(size_t)255; }

template<bool BIAS, bool GELU, bool RES, bool STF, bool STB>
static void launch_gemm(hipStream_t stream, int batches,
                        const __bf16* A, int lda, size_t sA,
                        const __bf16* Bt, int ldb, size_t sB,
                        int K, int Mvalid, int Nvalid,
                        const float* bias, const float* res, int ldr,
                        float* Cf, int ldc, size_t sC,
                        __bf16* Cb, int ldcb, size_t sCb)
{
    dim3 grid((Nvalid / 16 + 3) / 4, (Mvalid + 15) / 16, batches);
    gemm_wmma_kernel<BIAS, GELU, RES, STF, STB><<<grid, 32, 0, stream>>>(
        A, lda, sA, Bt, ldb, sB, K, Mvalid, Nvalid,
        bias, res, ldr, Cf, ldc, sC, Cb, ldcb, sCb);
}

extern "C" void kernel_launch(void* const* d_in, const int* in_sizes, int n_in,
                              void* d_out, int out_size, void* d_ws, size_t ws_size,
                              hipStream_t stream)
{
    (void)in_sizes; (void)n_in; (void)out_size; (void)ws_size;

    const float* x_in   = (const float*)d_in[0];
    const float* w_qkv  = (const float*)d_in[1];
    const float* b_qkv  = (const float*)d_in[2];
    const float* w_proj = (const float*)d_in[3];
    const float* b_proj = (const float*)d_in[4];
    const float* ln1_g  = (const float*)d_in[5];
    const float* ln1_b  = (const float*)d_in[6];
    const float* ln2_g  = (const float*)d_in[7];
    const float* ln2_b  = (const float*)d_in[8];
    const float* w_fc1  = (const float*)d_in[9];
    const float* b_fc1  = (const float*)d_in[10];
    const float* w_fc2  = (const float*)d_in[11];
    const float* b_fc2  = (const float*)d_in[12];
    const float* lnf_g  = (const float*)d_in[13];
    const float* lnf_b  = (const float*)d_in[14];

    float* outF    = (float*)d_out;                       // feats [B,N,D]
    float* outAttn = outF + (size_t)BATCH * NTOK * DIM;   // attn  [B,H,N,N]

    // ---- workspace carve (~137 MB total) ----
    char* ws = (char*)d_ws;
    size_t off = 0;
    auto alloc = [&](size_t bytes) -> void* {
        void* p = ws + off; off = align256(off + bytes); return p;
    };
    __bf16* wqkvT = (__bf16*)alloc((size_t)LAYERS * D3 * DIM * 2);
    __bf16* wprjT = (__bf16*)alloc((size_t)LAYERS * DIM * DIM * 2);
    __bf16* wfc1T = (__bf16*)alloc((size_t)LAYERS * D4 * DIM * 2);
    __bf16* wfc2T = (__bf16*)alloc((size_t)LAYERS * DIM * D4 * 2);
    float*  xcur  = (float*) alloc((size_t)MP * DIM * 4);
    float*  x2    = (float*) alloc((size_t)MP * DIM * 4);
    __bf16* xn    = (__bf16*)alloc((size_t)MP * DIM * 2);
    __bf16* hid   = (__bf16*)alloc((size_t)MP * D4 * 2);
    __bf16* attno = (__bf16*)alloc((size_t)MP * DIM * 2);
    float*  qkvf  = (float*) alloc((size_t)MP * D3 * 4);
    __bf16* Qb    = (__bf16*)alloc((size_t)BATCH * HEADS * NP * DH * 2);
    __bf16* Kb    = (__bf16*)alloc((size_t)BATCH * HEADS * NP * DH * 2);
    __bf16* VTb   = (__bf16*)alloc((size_t)BATCH * HEADS * DH * NP * 2);
    float*  scor  = (float*) alloc((size_t)BATCH * HEADS * NP * NP * 4);
    __bf16* attnb = (__bf16*)alloc((size_t)BATCH * HEADS * NP * NP * 2);

    // ---- weights f32 -> bf16 transposed (recomputed each call; deterministic)
    auto txp = [&](const float* W, int K, int N, __bf16* WT) {
        long long tot = (long long)K * N;
        transpose_to_bf16_kernel<<<(unsigned)((tot + 255) / 256), 256, 0, stream>>>(W, K, N, WT);
    };
    for (int l = 0; l < LAYERS; ++l) {
        txp(w_qkv + (size_t)l * DIM * D3, DIM, D3, wqkvT + (size_t)l * D3 * DIM);
        txp(w_proj + (size_t)l * DIM * DIM, DIM, DIM, wprjT + (size_t)l * DIM * DIM);
        txp(w_fc1 + (size_t)l * DIM * D4, DIM, D4, wfc1T + (size_t)l * D4 * DIM);
        txp(w_fc2 + (size_t)l * D4 * DIM, D4, DIM, wfc2T + (size_t)l * DIM * D4);
    }

    hipMemcpyAsync(xcur, x_in, (size_t)MROWS * DIM * 4, hipMemcpyDeviceToDevice, stream);

    const unsigned lnGrid = (MROWS + 3) / 4;
    const unsigned smGrid = (BATCH * HEADS * NTOK + 3) / 4;
    const unsigned upGrid = (BATCH * HEADS * NP * DH + 255) / 256;

    for (int l = 0; l < LAYERS; ++l) {
        const bool last = (l == LAYERS - 1);

        // ln1 -> xn (bf16)
        ln_kernel<false, true><<<lnGrid, 128, 0, stream>>>(
            xcur, MROWS, ln1_g + (size_t)l * DIM, ln1_b + (size_t)l * DIM, (float*)0, xn);

        // qkv = xn @ Wqkv + b  (f32)
        launch_gemm<true, false, false, true, false>(stream, 1,
            xn, DIM, 0, wqkvT + (size_t)l * D3 * DIM, DIM, 0,
            DIM, MROWS, D3, b_qkv + (size_t)l * D3, (const float*)0, 0,
            qkvf, D3, 0, (__bf16*)0, 0, 0);

        unpack_qkv_kernel<<<upGrid, 256, 0, stream>>>(qkvf, Qb, Kb, VTb);

        // scores[bh] = Q @ K^T   (batched over B*H; K operand IS Bt layout)
        launch_gemm<false, false, false, true, false>(stream, BATCH * HEADS,
            Qb, DH, (size_t)NP * DH, Kb, DH, (size_t)NP * DH,
            DH, NP, NP, (const float*)0, (const float*)0, 0,
            scor, NP, (size_t)NP * NP, (__bf16*)0, 0, 0);

        softmax_mask_kernel<<<smGrid, 128, 0, stream>>>(
            scor, attnb, last ? outAttn : (float*)0, BATCH * HEADS * NTOK);

        // attno = attn @ V  (bf16 out, head-interleaved columns; per-batch launch
        // so C batch stride is the uniform per-head 64-column offset)
        for (int b = 0; b < BATCH; ++b) {
            launch_gemm<false, false, false, false, true>(stream, HEADS,
                attnb + (size_t)b * HEADS * NP * NP, NP, (size_t)NP * NP,
                VTb + (size_t)b * HEADS * DH * NP, NP, (size_t)DH * NP,
                NP, NTOK, DH, (const float*)0, (const float*)0, 0,
                (float*)0, 0, 0, attno + (size_t)b * NTOK * DIM, DIM, DH);
        }

        // x2 = attno @ Wproj + b + xcur  (f32)
        launch_gemm<true, false, true, true, false>(stream, 1,
            attno, DIM, 0, wprjT + (size_t)l * DIM * DIM, DIM, 0,
            DIM, MROWS, DIM, b_proj + (size_t)l * DIM, xcur, DIM,
            x2, DIM, 0, (__bf16*)0, 0, 0);

        // ln2 -> xn
        ln_kernel<false, true><<<lnGrid, 128, 0, stream>>>(
            x2, MROWS, ln2_g + (size_t)l * DIM, ln2_b + (size_t)l * DIM, (float*)0, xn);

        // hid = gelu(xn @ Wfc1 + b)  (bf16)
        launch_gemm<true, true, false, false, true>(stream, 1,
            xn, DIM, 0, wfc1T + (size_t)l * D4 * DIM, DIM, 0,
            DIM, MROWS, D4, b_fc1 + (size_t)l * D4, (const float*)0, 0,
            (float*)0, 0, 0, hid, D4, 0);

        // xcur = hid @ Wfc2 + b + x2  (f32)
        launch_gemm<true, false, true, true, false>(stream, 1,
            hid, D4, 0, wfc2T + (size_t)l * DIM * D4, D4, 0,
            D4, MROWS, DIM, b_fc2 + (size_t)l * DIM, x2, DIM,
            xcur, DIM, 0, (__bf16*)0, 0, 0);
    }

    // final layernorm -> feats (f32, straight into d_out)
    ln_kernel<true, false><<<lnGrid, 128, 0, stream>>>(
        xcur, MROWS, lnf_g, lnf_b, outF, (__bf16*)0);
}
